// RadarPts_73074573574690
// MI455X (gfx1250) — matrix operand order repaired and checked by reading, not compiled, not run on hardware
//
#include <hip/hip_runtime.h>
#include <math.h>

// ---------------------------------------------------------------------------
// MI455X (gfx1250) implementation.
//   * All conv math: v_wmma_f32_16x16x32_bf16 (wave32), implicit GEMM over
//     NHWC bf16 activations, 16(pixel) x 32(out-ch) D tile per wave
//     (two accumulators sharing one A fragment).
//   * Exploits C % 32 == 0 (true for every conv in this network) so a K-step
//     of 32 never crosses an (r,s) boundary: A = two b128 vector loads,
//     B = one 32B contiguous load from [OC][Kdim]-layout weights.
//   * Pure 32-bit indexing everywhere (all tensors < 2^31 elements) and a
//     division-free carry-stepped epilogue: no 64-bit div/mul expansion.
//   * BN scale/bias, residual add and ReLU fused into the conv epilogue.
//   * Weight stream prefetched with global_prefetch (CDNA5 path).
// Workspace layout is carved linearly (~340 MB needed).
// ---------------------------------------------------------------------------

typedef __attribute__((ext_vector_type(16))) __bf16 v16bf;
typedef __attribute__((ext_vector_type(8)))  __bf16 v8bf;
typedef __attribute__((ext_vector_type(8)))  float  v8f;

// ============================ decode kernels ===============================

__global__ __launch_bounds__(256)
void k_nms_sig(const float* __restrict__ hm, float* __restrict__ heat,
               int BC, int H, int W) {
  unsigned i = blockIdx.x * 256u + threadIdx.x;
  unsigned tot = (unsigned)BC * H * W;
  if (i >= tot) return;
  int x = (int)(i % (unsigned)W);
  unsigned t = i / (unsigned)W;
  int y = (int)(t % (unsigned)H);
  int bc = (int)(t / (unsigned)H);
  const float* h = hm + (unsigned)bc * H * W;
  float v = h[y * W + x];
  float m = -1e38f;
  for (int dy = -1; dy <= 1; ++dy) {
    int yy = y + dy; if (yy < 0 || yy >= H) continue;
    for (int dx = -1; dx <= 1; ++dx) {
      int xx = x + dx; if (xx < 0 || xx >= W) continue;
      m = fmaxf(m, h[yy * W + xx]);
    }
  }
  // sigmoid is monotonic: NMS on logits == NMS on sigmoid(logits)
  heat[i] = (v == m) ? (1.f / (1.f + expf(-v))) : 0.f;
}

__global__ __launch_bounds__(256)
void k_topk_percls(float* __restrict__ heat, float* __restrict__ s1,
                   int* __restrict__ i1, int HW, int K) {
  int bc = blockIdx.x;
  float* h = heat + (unsigned)bc * HW;
  __shared__ float sv[256];
  __shared__ int   si[256];
  for (int k = 0; k < K; ++k) {
    float best = -1e38f; int bi = HW;
    for (int i = (int)threadIdx.x; i < HW; i += 256) {
      float v = h[i];
      if (v > best || (v == best && i < bi)) { best = v; bi = i; }
    }
    sv[threadIdx.x] = best; si[threadIdx.x] = bi;
    __syncthreads();
    for (int s = 128; s > 0; s >>= 1) {
      if ((int)threadIdx.x < s) {
        float ov = sv[threadIdx.x + s]; int oi = si[threadIdx.x + s];
        if (ov > sv[threadIdx.x] || (ov == sv[threadIdx.x] && oi < si[threadIdx.x])) {
          sv[threadIdx.x] = ov; si[threadIdx.x] = oi;
        }
      }
      __syncthreads();
    }
    if (threadIdx.x == 0) {
      s1[bc * K + k] = sv[0];
      i1[bc * K + k] = si[0];
      if (si[0] < HW) h[si[0]] = -1e30f;   // mark picked
    }
    __syncthreads();
  }
}

__global__ __launch_bounds__(256)
void k_topk_batch(const float* __restrict__ s1, const int* __restrict__ i1,
                  const float* __restrict__ wh, const float* __restrict__ reg,
                  float* __restrict__ dets, int C, int K, int H, int W) {
  int b = blockIdx.x;
  int n = C * K;                       // 300 candidates
  __shared__ float v[512];
  for (int i = (int)threadIdx.x; i < n; i += 256) v[i] = s1[b * n + i];
  __syncthreads();
  if (threadIdx.x != 0) return;
  int HW = H * W;
  for (int k = 0; k < K; ++k) {
    float best = -1e38f; int bj = 0;
    for (int j = 0; j < n; ++j)
      if (v[j] > best) { best = v[j]; bj = j; }  // strict >: first occurrence
    v[bj] = -1e30f;
    int cls = bj / K;
    int ind = i1[b * n + bj];
    float ys = (float)(ind / W);
    float xs = (float)(ind % W);
    float r0 = reg[(b * 2 + 0) * HW + ind];
    float r1 = reg[(b * 2 + 1) * HW + ind];
    float w0 = wh [(b * 2 + 0) * HW + ind];
    float w1 = wh [(b * 2 + 1) * HW + ind];
    float xc = xs + r0, yc = ys + r1;
    float* d = dets + (b * K + k) * 6;
    d[0] = xc - w0 * 0.5f; d[1] = yc - w1 * 0.5f;
    d[2] = xc + w0 * 0.5f; d[3] = yc + w1 * 0.5f;
    d[4] = best; d[5] = (float)cls;
  }
}

__global__ __launch_bounds__(256)
void k_detmask(const float* __restrict__ dets, float* __restrict__ msk,
               int B, int H, int W) {
  unsigned i = blockIdx.x * 256u + threadIdx.x;
  unsigned tot = (unsigned)B * H * W;
  if (i >= tot) return;
  int x = (int)(i % (unsigned)W);
  unsigned t = i / (unsigned)W;
  int y = (int)(t % (unsigned)H);
  int b = (int)(t / (unsigned)H);
  float out = 0.f;
  for (int j = 0; j < 20; ++j) {
    const float* d = dets + (b * 100 + j) * 6;
    if (d[4] >= 0.3f) {
      int x1 = (int)d[0], y1 = (int)d[1], x2 = (int)d[2], y2 = (int)d[3];
      if (y >= y1 && y < y2 && x >= x1 && x < x2) { out = 1.f; break; }
    }
  }
  msk[i] = out;
}

// mask-multiply + NCHW f32 -> NHWC bf16
__global__ __launch_bounds__(256)
void k_maskcvt(const float* __restrict__ dif, const float* __restrict__ msk,
               __bf16* __restrict__ out, int B, int C, int H, int W) {
  unsigned i = blockIdx.x * 256u + threadIdx.x;
  unsigned tot = (unsigned)B * H * W * C;
  if (i >= tot) return;
  int c = (int)(i % (unsigned)C);
  unsigned pix = i / (unsigned)C;
  int x = (int)(pix % (unsigned)W);
  unsigned t = pix / (unsigned)W;
  int y = (int)(t % (unsigned)H);
  int b = (int)(t / (unsigned)H);
  float m = msk[pix];                       // (b*H + y)*W + x == pix
  float v = dif[((b * C + c) * H + y) * W + x] * m;
  out[i] = (__bf16)v;
}

// ============================ helpers ======================================

// NHWC channel-slice copy (also plain copy when srcC==dstC, off==0)
__global__ __launch_bounds__(256)
void k_copychan(const __bf16* __restrict__ src, int srcC,
                __bf16* __restrict__ dst, int dstC, int off, int pixels) {
  unsigned i = blockIdx.x * 256u + threadIdx.x;
  unsigned tot = (unsigned)pixels * srcC;
  if (i >= tot) return;
  int c = (int)(i % (unsigned)srcC);
  unsigned p = i / (unsigned)srcC;
  dst[p * dstC + off + c] = src[i];
}

// align-corners bilinear (matches jnp.linspace(0, h-1, oh)) on NHWC bf16
__global__ __launch_bounds__(256)
void k_resize_nhwc(const __bf16* __restrict__ in, __bf16* __restrict__ out,
                   int N, int H, int W, int C, int OH, int OW) {
  unsigned i = blockIdx.x * 256u + threadIdx.x;
  unsigned tot = (unsigned)N * OH * OW * C;
  if (i >= tot) return;
  int c = (int)(i % (unsigned)C);
  unsigned t = i / (unsigned)C;
  int ox = (int)(t % (unsigned)OW); t /= (unsigned)OW;
  int oy = (int)(t % (unsigned)OH);
  int n  = (int)(t / (unsigned)OH);
  float fy = (OH > 1) ? (float)oy * (float)(H - 1) / (float)(OH - 1) : 0.f;
  float fx = (OW > 1) ? (float)ox * (float)(W - 1) / (float)(OW - 1) : 0.f;
  int y0 = (int)floorf(fy); int y1 = (y0 + 1 < H) ? y0 + 1 : H - 1;
  int x0 = (int)floorf(fx); int x1 = (x0 + 1 < W) ? x0 + 1 : W - 1;
  float wy = fy - (float)y0, wx = fx - (float)x0;
  const int strH = W * C;
  const __bf16* base = in + (unsigned)n * H * strH + c;
  float a  = (float)base[y0 * strH + x0 * C];
  float bq = (float)base[y0 * strH + x1 * C];
  float cc = (float)base[y1 * strH + x0 * C];
  float d  = (float)base[y1 * strH + x1 * C];
  float top = a  * (1.f - wx) + bq * wx;
  float bot = cc * (1.f - wx) + d  * wx;
  out[i] = (__bf16)(top * (1.f - wy) + bot * wy);
}

// fold BN into per-channel scale/bias (or bias-only / identity)
__global__ __launch_bounds__(256)
void k_fusebn(const float* g, const float* b, const float* m, const float* v,
              const float* biasOnly, float* __restrict__ sc,
              float* __restrict__ bi, int n) {
  int i = blockIdx.x * 256 + threadIdx.x;
  if (i >= n) return;
  if (g) {
    float s = g[i] * rsqrtf(v[i] + 1e-5f);
    sc[i] = s;
    bi[i] = b[i] - m[i] * s;
  } else {
    sc[i] = 1.f;
    bi[i] = biasOnly ? biasOnly[i] : 0.f;
  }
}

// OIHW f32 -> [OC][Kdim] bf16, Kdim index = (r*S+s)*C + c.
// 3-D grid (C-chunks, R*S, OC): no per-element div/mod chains.
__global__ __launch_bounds__(256)
void k_wcvt(const float* __restrict__ w, __bf16* __restrict__ wb,
            int C, int R, int S) {
  int c  = (int)(blockIdx.x * 256u + threadIdx.x);
  if (c >= C) return;
  int rs = (int)blockIdx.y;
  int oc = (int)blockIdx.z;
  int r = rs / S;
  int s = rs - r * S;
  int Kdim = R * S * C;
  wb[oc * Kdim + rs * C + c] = (__bf16)w[((oc * C + c) * R + r) * S + s];
}

// ===================== implicit-GEMM WMMA conv =============================
// One wave per block; wave owns a 16(pixel) x 32(out-ch) D tile (2 acc).
// REQUIRES C % 32 == 0 (holds for every conv in this network), so each
// K-step of 32 stays inside one (r,s) tap:
//   A (16x32 bf16): lane row = pixel; elem e = c0+lhalf*16+e
//                   -> two b128 loads.
//   B (32x16 bf16): lane col = oc; elem e = K = lhalf*16+e -> one 32B load.
//   D (16x16 f32):  col = lane&15; acc[v] is row v + 8*(lane>>4).
// All indexing is 32-bit; epilogue coords are carry-stepped (no division).
__global__ __launch_bounds__(32)
void k_conv_wmma(const __bf16* __restrict__ act, const __bf16* __restrict__ wgt,
                 const float* __restrict__ scale, const float* __restrict__ bias,
                 const __bf16* __restrict__ resid, __bf16* __restrict__ out,
                 float* __restrict__ foutNCHW,
                 int N, int H, int W, int C, int OC, int R, int S,
                 int stride, int pad, int P, int Q, int relu) {
  const int lane  = (int)threadIdx.x;
  const int lhalf = lane >> 4;
  const int lrow  = lane & 15;
  const int tileN = (int)blockIdx.x;   // 32-wide OC tile
  const int tileM = (int)blockIdx.y;   // 16-wide pixel tile
  const int PQ   = P * Q;
  const int M    = N * PQ;
  const int Kdim = R * S * C;

  // pixel owned by this lane for A-loading (one 32-bit div/mod pair)
  int gm = tileM * 16 + lrow;
  bool mvalid = gm < M;
  int b = 0, p = 0, q = 0;
  if (mvalid) {
    b = gm / PQ; int rpq = gm - b * PQ;
    p = rpq / Q; q = rpq - p * Q;
  }
  const int oc0 = tileN * 32 + lrow;
  const int oc1 = oc0 + 16;
  const bool bv0 = oc0 < OC;
  const bool bv1 = oc1 < OC;
  const __bf16* bp0 = wgt + oc0 * Kdim + lhalf * 16;
  const __bf16* bp1 = wgt + oc1 * Kdim + lhalf * 16;

  const v8bf  z8  = {};
  const v16bf z16 = {};
  v8f acc0 = {};
  v8f acc1 = {};

  int kpos = 0;
  for (int r = 0; r < R; ++r) {
    const int ih = p * stride - pad + r;
    const bool hv = mvalid && ih >= 0 && ih < H;
    const int cih = hv ? ih : 0;
    for (int s = 0; s < S; ++s) {
      const int iw = q * stride - pad + s;
      const bool av = hv && iw >= 0 && iw < W;
      const int ciw = av ? iw : 0;
      const __bf16* aptr = act + ((b * H + cih) * W + ciw) * C + lhalf * 16;
      for (int c0 = 0; c0 < C; c0 += 32, kpos += 32) {
        v8bf lo = av ? *(const v8bf*)(aptr + c0)      : z8;
        v8bf hi = av ? *(const v8bf*)(aptr + c0 + 8) : z8;
        v16bf va = __builtin_shufflevector(
            lo, hi, 0, 1, 2, 3, 4, 5, 6, 7, 8, 9, 10, 11, 12, 13, 14, 15);
        v16bf vb0 = bv0 ? *(const v16bf*)(bp0 + kpos) : z16;
        v16bf vb1 = bv1 ? *(const v16bf*)(bp1 + kpos) : z16;
        if (kpos + 32 < Kdim) {  // stream next weight panel (global_prefetch)
          __builtin_prefetch(bp0 + kpos + 32, 0, 1);
          __builtin_prefetch(bp1 + kpos + 32, 0, 1);
        }
        acc0 = __builtin_amdgcn_wmma_f32_16x16x32_bf16(
            false, va, false, vb0, (short)0, acc0, false, false);
        acc1 = __builtin_amdgcn_wmma_f32_16x16x32_bf16(
            false, va, false, vb1, (short)0, acc1, false, false);
      }
    }
  }

  // ---- epilogue: coords of this lane's 8 output rows, carry-stepped ----
  int m0 = tileM * 16 + 8 * lhalf;
  int be = 0, pe = 0, qe = 0;
  if (m0 < M) {
    be = m0 / PQ; int rpq = m0 - be * PQ;
    pe = rpq / Q; qe = rpq - pe * Q;
  }
  bool mok[8]; int bea[8], pea[8], qea[8];
#pragma unroll
  for (int v = 0; v < 8; ++v) {
    mok[v] = (m0 + v) < M;
    bea[v] = be; pea[v] = pe; qea[v] = qe;
    if (++qe == Q) { qe = 0; if (++pe == P) { pe = 0; ++be; } }
  }

  auto store_tile = [&](int ocB, const v8f& acc) {
    if (ocB >= OC) return;
    float sc = scale[ocB], bs = bias[ocB];
#pragma unroll
    for (int v = 0; v < 8; ++v) {
      if (!mok[v]) continue;
      int pix = (bea[v] * P + pea[v]) * Q + qea[v];
      float r = acc[v] * sc + bs;
      if (resid) r += (float)resid[pix * OC + ocB];
      if (relu)  r = fmaxf(r, 0.f);
      if (foutNCHW)
        foutNCHW[((bea[v] * OC + ocB) * P + pea[v]) * Q + qea[v]] = r;
      else
        out[pix * OC + ocB] = (__bf16)r;
    }
  };
  store_tile(oc0, acc0);
  store_tile(oc1, acc1);
}

// ============================ host driver ==================================

struct BNp  { const float *g, *b, *m, *v; };
struct Blk  { BNp bn1, bn2; const float *c1, *c2; bool hasDown; const float *dw; BNp dbn; };
struct Head { BNp b1, b2, b3; const float *c1, *c2, *c3, *c4, *c4b; };
struct Up   { BNp b1, b2; const float *c1, *c2; };

extern "C" void kernel_launch(void* const* d_in, const int* in_sizes, int n_in,
                              void* d_out, int out_size, void* d_ws, size_t ws_size,
                              hipStream_t stream) {
  (void)in_sizes; (void)n_in; (void)out_size; (void)ws_size;

  const int B = 2, CCLS = 3, H = 300, W = 400, CIN = 32;

  const float* dif = (const float*)d_in[0];
  const float* hm  = (const float*)d_in[1];
  const float* wh  = (const float*)d_in[2];
  const float* reg = (const float*)d_in[3];

  // ---- params: jax tree-flatten order (nested dicts sorted by key) ----
  int cur = 4;
  auto nextF  = [&]() { return (const float*)d_in[cur++]; };
  auto nextBN = [&]() { BNp p; p.b = nextF(); p.g = nextF(); p.m = nextF(); p.v = nextF(); return p; };
  auto nextBlk = [&](bool down) {
    Blk k; k.bn1 = nextBN(); k.bn2 = nextBN(); k.c1 = nextF(); k.c2 = nextF();
    k.hasDown = down; k.dw = nullptr;
    if (down) { k.dbn = nextBN(); k.dw = nextF(); }   // 'dbn' < 'down'
    return k;
  };
  auto nextHead = [&]() {
    Head h; h.b1 = nextBN(); h.b2 = nextBN(); h.b3 = nextBN();
    h.c1 = nextF(); h.c2 = nextF(); h.c3 = nextF(); h.c4 = nextF(); h.c4b = nextF();
    return h;
  };
  auto nextUp = [&]() { Up u; u.b1 = nextBN(); u.b2 = nextBN(); u.c1 = nextF(); u.c2 = nextF(); return u; };

  BNp bn1 = nextBN();                 // 'bn1'
  const float* conv1w = nextF();      // 'conv1'
  Head htHead = nextHead();           // 'ht_head'
  Blk L1[3]; for (int i = 0; i < 3; ++i) L1[i] = nextBlk(false);        // 'layer1'
  Blk L2[4]; L2[0] = nextBlk(true); for (int i = 1; i < 4; ++i) L2[i] = nextBlk(false);
  Blk L3[6]; L3[0] = nextBlk(true); for (int i = 1; i < 6; ++i) L3[i] = nextBlk(false);
  Head mskHead = nextHead();          // 'msk_head'
  Up up1   = nextUp();                // 'up1'
  Up up2_1 = nextUp();                // 'up2_1'
  Up up2_2 = nextUp();                // 'up2_2'

  // ---- workspace carve (~340 MB) ----
  size_t off = 0;
  auto carve = [&](size_t bytes) -> void* {
    void* p = (char*)d_ws + off;
    off += (bytes + 255) & ~(size_t)255;
    return p;
  };
  float* heat   = (float*)carve((size_t)B * CCLS * H * W * 4);
  float* s1     = (float*)carve((size_t)B * CCLS * 100 * 4);
  int*   i1     = (int*)  carve((size_t)B * CCLS * 100 * 4);
  float* dets   = (float*)carve((size_t)B * 100 * 6 * 4);
  float* dScale = (float*)carve(1024);
  float* dBias  = (float*)carve(1024);
  __bf16* dWb   = (__bf16*)carve((size_t)9 * 320 * 256 * 2);
  __bf16* big0  = (__bf16*)carve((size_t)B * 300 * 400 * 256 * 2);
  __bf16* big1  = (__bf16*)carve((size_t)B * 300 * 400 * 256 * 2);
  __bf16* bufT  = (__bf16*)carve((size_t)B * 150 * 200 * 64 * 2);
  __bf16* bufS  = (__bf16*)carve((size_t)B * 75 * 100 * 128 * 2);
  __bf16* bufX1 = (__bf16*)carve((size_t)B * 150 * 200 * 64 * 2);
  __bf16* bufCat= (__bf16*)carve((size_t)B * 150 * 200 * 320 * 2);
  __bf16* bufUp = (__bf16*)carve((size_t)B * 150 * 200 * 256 * 2);

  float* outMsk = (float*)d_out;                       // (2,2,300,400)
  float* outHt  = (float*)d_out + (size_t)B * 2 * H * W;
  float* outDm  = (float*)d_out + (size_t)2 * B * 2 * H * W;  // (2,1,300,400)

  // ---- decode ----
  {
    unsigned tot = (unsigned)B * CCLS * H * W;
    k_nms_sig<<<dim3((tot + 255) / 256), 256, 0, stream>>>(hm, heat, B * CCLS, H, W);
    k_topk_percls<<<dim3(B * CCLS), 256, 0, stream>>>(heat, s1, i1, H * W, 100);
    k_topk_batch<<<dim3(B), 256, 0, stream>>>(s1, i1, wh, reg, dets, CCLS, 100, H, W);
    unsigned tm = (unsigned)B * H * W;
    k_detmask<<<dim3((tm + 255) / 256), 256, 0, stream>>>(dets, outDm, B, H, W);
    unsigned tc = (unsigned)B * H * W * CIN;
    k_maskcvt<<<dim3((tc + 255) / 256), 256, 0, stream>>>(dif, outDm, big0, B, CIN, H, W);
  }

  // ---- conv driver ----
  auto runConv = [&](const __bf16* in, int Hh, int Ww, int C,
                     const float* w, int OC, int R, int S, int stride, int pad,
                     const BNp* bn, const float* biasOnly, int relu,
                     const __bf16* resid, __bf16* outp, float* foutNCHW,
                     int& P, int& Q) {
    P = (Hh + 2 * pad - R) / stride + 1;
    Q = (Ww + 2 * pad - S) / stride + 1;
    k_fusebn<<<dim3((unsigned)((OC + 255) / 256)), 256, 0, stream>>>(
        bn ? bn->g : nullptr, bn ? bn->b : nullptr, bn ? bn->m : nullptr,
        bn ? bn->v : nullptr, biasOnly, dScale, dBias, OC);
    dim3 wg((unsigned)((C + 255) / 256), (unsigned)(R * S), (unsigned)OC);
    k_wcvt<<<wg, 256, 0, stream>>>(w, dWb, C, R, S);
    dim3 grid((unsigned)((OC + 31) / 32), (unsigned)((B * P * Q + 15) / 16));
    k_conv_wmma<<<grid, 32, 0, stream>>>(in, dWb, dScale, dBias, resid, outp, foutNCHW,
                                         B, Hh, Ww, C, OC, R, S, stride, pad, P, Q, relu);
  };

  auto runBlock = [&](const __bf16* X, int Hh, int Ww, int C, const Blk& bk,
                      int strideB, int OC, __bf16* outbuf, int& P, int& Q) {
    int p1, q1;
    runConv(X, Hh, Ww, C, bk.c1, OC, 3, 3, strideB, 1, &bk.bn1, nullptr, 1,
            nullptr, bufT, nullptr, p1, q1);
    const __bf16* sc = X;
    if (bk.hasDown) {
      int pd, qd;
      runConv(X, Hh, Ww, C, bk.dw, OC, 1, 1, strideB, 0, &bk.dbn, nullptr, 0,
              nullptr, bufS, nullptr, pd, qd);
      sc = bufS;
    }
    runConv(bufT, p1, q1, OC, bk.c2, OC, 3, 3, 1, 1, &bk.bn2, nullptr, 1,
            sc, outbuf, nullptr, P, Q);
  };

  int P, Q;

  // stem: 7x7 s2 p3, 32->64, bn1+relu : big0 -> big1  (150x200)
  runConv(big0, H, W, CIN, conv1w, 64, 7, 7, 2, 3, &bn1, nullptr, 1,
          nullptr, big1, nullptr, P, Q);

  __bf16* X = big1; __bf16* Y = big0;
  auto flip = [&]() { __bf16* t = X; X = Y; Y = t; };

  // layer1: 3 blocks, 64ch @150x200
  for (int i = 0; i < 3; ++i) { runBlock(X, 150, 200, 64, L1[i], 1, 64, Y, P, Q); flip(); }
  {
    int pix = B * 150 * 200 * 64;   // elementwise copy of x1 skip
    k_copychan<<<dim3((unsigned)((pix + 255) / 256)), 256, 0, stream>>>(X, 1, bufX1, 1, 0, pix);
  }

  // layer2: 128ch, first stride 2 -> 75x100
  runBlock(X, 150, 200, 64, L2[0], 2, 128, Y, P, Q); flip();
  for (int i = 1; i < 4; ++i) { runBlock(X, 75, 100, 128, L2[i], 1, 128, Y, P, Q); flip(); }

  // layer3: 256ch, first stride 2 -> 38x50
  runBlock(X, 75, 100, 128, L3[0], 2, 256, Y, P, Q); flip();
  for (int i = 1; i < 6; ++i) { runBlock(X, 38, 50, 256, L3[i], 1, 256, Y, P, Q); flip(); }

  // resize 38x50 -> 150x200 (align corners), concat [x1(64) | up(256)] -> 320ch
  {
    unsigned tot = (unsigned)B * 150 * 200 * 256;
    k_resize_nhwc<<<dim3((tot + 255) / 256), 256, 0, stream>>>(
        X, Y, B, 38, 50, 256, 150, 200);
    int pix = B * 150 * 200;
    k_copychan<<<dim3((unsigned)((pix * 64 + 255) / 256)), 256, 0, stream>>>(
        bufX1, 64, bufCat, 320, 0, pix);
    k_copychan<<<dim3((unsigned)((pix * 256 + 255) / 256)), 256, 0, stream>>>(
        Y, 256, bufCat, 320, 64, pix);
  }

  // up1: 320->256, 256->256 (both cbr) -> bufUp (persist for both branches)
  runConv(bufCat, 150, 200, 320, up1.c1, 256, 3, 3, 1, 1, &up1.b1, nullptr, 1,
          nullptr, big0, nullptr, P, Q);
  runConv(big0, 150, 200, 256, up1.c2, 256, 3, 3, 1, 1, &up1.b2, nullptr, 1,
          nullptr, bufUp, nullptr, P, Q);

  auto runHead = [&](const __bf16* in, const Head& hd, float* fout) {
    int p, q;
    runConv(in,   300, 400, 64,  hd.c1, 256, 3, 3, 1, 1, &hd.b1, nullptr, 1, nullptr, big1, nullptr, p, q);
    runConv(big1, 300, 400, 256, hd.c2, 128, 3, 3, 1, 1, &hd.b2, nullptr, 1, nullptr, big0, nullptr, p, q);
    runConv(big0, 300, 400, 128, hd.c3, 32,  3, 3, 1, 1, &hd.b3, nullptr, 1, nullptr, big1, nullptr, p, q);
    runConv(big1, 300, 400, 32,  hd.c4, 2,   1, 1, 1, 0, nullptr, hd.c4b, 0, nullptr, nullptr, fout, p, q);
  };

  auto runBranch = [&](const Up& u, const Head& hd, float* fout) {
    unsigned tot = (unsigned)B * 300 * 400 * 256;
    k_resize_nhwc<<<dim3((tot + 255) / 256), 256, 0, stream>>>(
        bufUp, big0, B, 150, 200, 256, 300, 400);
    int p, q;
    runConv(big0, 300, 400, 256, u.c1, 128, 3, 3, 1, 1, &u.b1, nullptr, 1, nullptr, big1, nullptr, p, q);
    runConv(big1, 300, 400, 128, u.c2, 64,  3, 3, 1, 1, &u.b2, nullptr, 1, nullptr, big0, nullptr, p, q);
    runHead(big0, hd, fout);
  };

  runBranch(up2_1, mskHead, outMsk);
  runBranch(up2_2, htHead,  outHt);
}
